// DynamicsAgent_4947802325750
// MI455X (gfx1250) — matrix operand order repaired and checked by reading, not compiled, not run on hardware
//
#include <hip/hip_runtime.h>
#include <hip/hip_bf16.h>
#include <math.h>

typedef __attribute__((ext_vector_type(16))) __bf16 v16bf;
typedef __attribute__((ext_vector_type(8)))  __bf16 v8bf;
typedef __attribute__((ext_vector_type(8)))  float  v8f;

union BF16x16 { v16bf v; v8bf h[2]; };

#define NTRY  100000
#define TSTEPS 12
#define HDIM  200
#define SDIM  30
#define ADIM  6
#define XCOLS 256
#define NKEEP 100
#define NTILES (NTRY / 16)          // 6250
#define WPB   2                     // waves per block (64 threads)

#define RSSM_N 208
#define HEAD_N 32
#define REW_N  16

static __device__ __forceinline__ v8f wmma_bf16(const BF16x16& a, const BF16x16& b, v8f c) {
  return __builtin_amdgcn_wmma_f32_16x16x32_bf16(false, a.v, false, b.v, (short)0, c, false, false);
}

// ---- CDNA5 async global->LDS copy (ASYNCcnt-tracked) ----
static __device__ __forceinline__ void async_copy_b128(unsigned lds_off, const void* gaddr) {
  asm volatile("global_load_async_to_lds_b128 %0, %1, off"
               :: "v"(lds_off), "v"((unsigned long long)gaddr) : "memory");
}
static __device__ __forceinline__ void wait_async0() {
  asm volatile("s_wait_asynccnt 0" ::: "memory");
}
static __device__ __forceinline__ unsigned lds_off_of(const void* p) {
  return (unsigned)(uintptr_t)p;   // low 32 bits of generic ptr = addrspace(3) offset
}

static __device__ __forceinline__ float tanh_fast(float x) {
  float ax = fabsf(x);
  float e  = __expf(-2.0f * ax);
  float t  = (1.0f - e) * __builtin_amdgcn_rcpf(1.0f + e);
  return copysignf(t, x);
}
static __device__ __forceinline__ float softplus_fast(float x) {
  return (x > 20.0f) ? x : __logf(1.0f + __expf(x));
}

// ---------------------------------------------------------------------------
// Kernel 1: convert weights to zero-padded bf16, K-major layout.
// ---------------------------------------------------------------------------
__global__ void prep_weights(const float* __restrict__ W_rssm,
                             const float* __restrict__ W_mu,
                             const float* __restrict__ W_sig,
                             const float* __restrict__ W_rew,
                             __bf16* __restrict__ Wb_rssm,
                             __bf16* __restrict__ Wb_mu,
                             __bf16* __restrict__ Wb_sig,
                             __bf16* __restrict__ Wb_rew) {
  int i = blockIdx.x * blockDim.x + threadIdx.x;
  if (i < XCOLS * RSSM_N) {
    int n = i / XCOLS, k = i % XCOLS;
    float v = (k < HDIM + SDIM + ADIM && n < HDIM) ? W_rssm[k * HDIM + n] : 0.0f;
    Wb_rssm[i] = (__bf16)v;
  }
  int j = i - XCOLS * RSSM_N;
  if (j >= 0 && j < XCOLS * HEAD_N) {
    int n = j / XCOLS, k = j % XCOLS;
    bool ok = (k < HDIM && n < SDIM);
    Wb_mu[j]  = (__bf16)(ok ? W_mu[k * SDIM + n]  : 0.0f);
    Wb_sig[j] = (__bf16)(ok ? W_sig[k * SDIM + n] : 0.0f);
  }
  int m = i - XCOLS * RSSM_N - XCOLS * HEAD_N;
  if (m >= 0 && m < XCOLS * REW_N) {
    int n = m / XCOLS, k = m % XCOLS;
    float v = (k < HDIM + SDIM && n == 0) ? W_rew[k] : 0.0f;
    Wb_rew[m] = (__bf16)v;
  }
}

// ---------------------------------------------------------------------------
// Kernel 2: full T-step rollout. One wave per 16-trajectory tile.
// Ping-pong x buffers + double-buffered async eps staging.
// ---------------------------------------------------------------------------
__global__ __launch_bounds__(WPB * 32) void rollout_kernel(
    const float* __restrict__ h0, const float* __restrict__ s0,
    const float* __restrict__ act_mus, const float* __restrict__ act_sigmas,
    const float* __restrict__ eps_a, const float* __restrict__ eps_s,
    const float* __restrict__ b_rssm, const float* __restrict__ b_mu,
    const float* __restrict__ b_sig, const float* __restrict__ b_rew,
    const __bf16* __restrict__ Wb_rssm, const __bf16* __restrict__ Wb_mu,
    const __bf16* __restrict__ Wb_sig, const __bf16* __restrict__ Wb_rew,
    float* __restrict__ rew_out) {
  __shared__ __align__(16) __bf16 xbuf_all[WPB][2][16][XCOLS];    // 2*16 KB
  __shared__ __align__(16) float  epsS_all[WPB][2][16 * SDIM];    // 2*3.75 KB
  __shared__ __align__(16) float  epsA_all[WPB][2][16 * ADIM];    // 2*0.75 KB

  const int wave = threadIdx.x >> 5;
  const int lane = threadIdx.x & 31;
  const int tile = blockIdx.x * WPB + wave;
  if (tile >= NTILES) return;
  const int row0 = tile * 16;

  __bf16 (*xb)[16][XCOLS] = xbuf_all[wave];
  float (*eS)[16 * SDIM] = epsS_all[wave];
  float (*eA)[16 * ADIM] = epsA_all[wave];

  // Init x buffers: buf 0 = [h0 | s0 | 0], buf 1 = 0; pad cols stay zero forever.
  for (int idx = lane; idx < 2 * 16 * XCOLS; idx += 32) {
    int b = idx >> 12, rc = idx & 4095;
    int r = rc >> 8, c = rc & (XCOLS - 1);
    float v = 0.0f;
    if (b == 0) {
      if (c < HDIM)             v = h0[(size_t)(row0 + r) * HDIM + c];
      else if (c < HDIM + SDIM) v = s0[(size_t)(row0 + r) * SDIM + (c - HDIM)];
    }
    xb[b][r][c] = (__bf16)v;
  }

  // Prologue: async-stage eps tiles for t=0 into buffer 0.
  {
    const char* gs = (const char*)(eps_s + (size_t)row0 * SDIM);
    unsigned ls = lds_off_of(&eS[0][0]);
    for (int c = lane; c < (16 * SDIM * 4) / 16; c += 32)   // 120 x b128
      async_copy_b128(ls + c * 16, gs + c * 16);
    const char* ga = (const char*)(eps_a + (size_t)row0 * ADIM);
    unsigned la = lds_off_of(&eA[0][0]);
    for (int c = lane; c < (16 * ADIM * 4) / 16; c += 32)   // 24 x b128
      async_copy_b128(la + c * 16, ga + c * 16);
  }

  const int arow = lane & 15;
  const int ab   = (lane >> 4) << 3;    // A-frag K sub-offset (0 or 8)
  const int cl   = lane & 15;           // C/D column within tile
  const int rb   = (lane >> 4) << 3;    // C/D row base (0 or 8)
  const float brew = b_rew[0];

  v8f rew_acc = {};                     // col 0 lives in lanes 0 and 16

  for (int t = 0; t < TSTEPS; ++t) {
    const int cur = t & 1, nxt = cur ^ 1;
    __bf16 (*xc)[XCOLS] = xb[cur];      // holds h_{t-1}, s_{t-1}
    __bf16 (*xn)[XCOLS] = xb[nxt];      // will hold h_t, s_t

    wait_async0();                      // eps tiles for step t are in LDS

    // actions a_t = mu + sigma*eps_a -> x_cur cols 230..235
    for (int idx = lane; idx < 16 * ADIM; idx += 32) {
      int r = idx / ADIM, c = idx % ADIM;
      float a = act_mus[t * ADIM + c] +
                act_sigmas[t * ADIM + c] * eA[cur][idx];
      xc[r][HDIM + SDIM + c] = (__bf16)a;
    }

    // Prefetch eps tiles for step t+1 (overlaps with WMMA below).
    if (t + 1 < TSTEPS) {
      const char* gs = (const char*)(eps_s + ((size_t)(t + 1) * NTRY + row0) * SDIM);
      unsigned ls = lds_off_of(&eS[nxt][0]);
      for (int c = lane; c < (16 * SDIM * 4) / 16; c += 32)
        async_copy_b128(ls + c * 16, gs + c * 16);
      const char* ga = (const char*)(eps_a + ((size_t)(t + 1) * NTRY + row0) * ADIM);
      unsigned la = lds_off_of(&eA[nxt][0]);
      for (int c = lane; c < (16 * ADIM * 4) / 16; c += 32)
        async_copy_b128(la + c * 16, ga + c * 16);
    }

    // --- h_t = tanh(x @ W_rssm + b): 13 N-tiles in 2 passes (reg pressure) ---
#pragma unroll
    for (int pass = 0; pass < 2; ++pass) {
      const int nt0 = pass ? 7 : 0;
      const int ntn = pass ? 6 : 7;
      v8f acc[7];
#pragma unroll
      for (int q = 0; q < 7; ++q) acc[q] = (v8f){};
      for (int ks = 0; ks < 8; ++ks) {
        const int k0 = ks * 32;
        BF16x16 af;
        af.h[0] = *(const v8bf*)&xc[arow][k0 + ab];
        af.h[1] = *(const v8bf*)&xc[arow][k0 + ab + 16];
        const __bf16* wrow = Wb_rssm + (size_t)(nt0 * 16 + arow) * XCOLS + k0 + ab;
#pragma unroll
        for (int q = 0; q < 7; ++q) {
          if (q < ntn) {
            BF16x16 bf;
            bf.h[0] = *(const v8bf*)(wrow + q * 16 * XCOLS);
            bf.h[1] = *(const v8bf*)(wrow + q * 16 * XCOLS + 16);
            acc[q] = wmma_bf16(af, bf, acc[q]);
          }
        }
      }
#pragma unroll
      for (int q = 0; q < 7; ++q) {
        if (q < ntn) {
          int c = (nt0 + q) * 16 + cl;
          if (c < HDIM) {
            float bias = b_rssm[c];
#pragma unroll
            for (int i = 0; i < 8; ++i)
              xn[rb + i][c] = (__bf16)tanh_fast(acc[q][i] + bias);
          }
        }
      }
    }

    // --- heads on x_next (K=256 padded; W rows k>=200 are zero) ---
    v8f am[2], as2[2];
    am[0] = am[1] = as2[0] = as2[1] = (v8f){};
    for (int ks = 0; ks < 8; ++ks) {
      const int k0 = ks * 32;
      BF16x16 af;
      af.h[0] = *(const v8bf*)&xn[arow][k0 + ab];
      af.h[1] = *(const v8bf*)&xn[arow][k0 + ab + 16];
      const __bf16* wm = Wb_mu  + (size_t)arow * XCOLS + k0 + ab;
      const __bf16* ws = Wb_sig + (size_t)arow * XCOLS + k0 + ab;
#pragma unroll
      for (int nt = 0; nt < 2; ++nt) {
        BF16x16 bm, bs;
        bm.h[0] = *(const v8bf*)(wm + nt * 16 * XCOLS); bm.h[1] = *(const v8bf*)(wm + nt * 16 * XCOLS + 16);
        bs.h[0] = *(const v8bf*)(ws + nt * 16 * XCOLS); bs.h[1] = *(const v8bf*)(ws + nt * 16 * XCOLS + 16);
        am[nt]  = wmma_bf16(af, bm, am[nt]);
        as2[nt] = wmma_bf16(af, bs, as2[nt]);
      }
    }
#pragma unroll
    for (int nt = 0; nt < 2; ++nt) {
      int c = nt * 16 + cl;
      if (c < SDIM) {
        float bmu = b_mu[c], bsg = b_sig[c];
#pragma unroll
        for (int i = 0; i < 8; ++i) {
          int r = rb + i;
          float mu = am[nt][i] + bmu;
          float sp = softplus_fast(as2[nt][i] + bsg);
          float es = eS[cur][r * SDIM + c];
          xn[r][HDIM + c] = (__bf16)(mu + sp * es);
        }
      }
    }

    // --- rew = [h_t|s_t] @ W_rew + b_rew (W rows k>=230 zero) ---
    v8f racc = {};
    for (int ks = 0; ks < 8; ++ks) {
      const int k0 = ks * 32;
      BF16x16 af, bf;
      af.h[0] = *(const v8bf*)&xn[arow][k0 + ab];
      af.h[1] = *(const v8bf*)&xn[arow][k0 + ab + 16];
      const __bf16* wr = Wb_rew + (size_t)arow * XCOLS + k0 + ab;
      bf.h[0] = *(const v8bf*)(wr);
      bf.h[1] = *(const v8bf*)(wr + 16);
      racc = wmma_bf16(af, bf, racc);
    }
#pragma unroll
    for (int i = 0; i < 8; ++i) rew_acc[i] += racc[i] + brew;
  }

  if ((lane & 15) == 0) {
#pragma unroll
    for (int i = 0; i < 8; ++i) rew_out[row0 + rb + i] = rew_acc[i];
  }
}

// ---------------------------------------------------------------------------
// Kernel 3: iterative top-k extraction (destructive on rew buffer).
// ---------------------------------------------------------------------------
__global__ __launch_bounds__(512) void topk_kernel(float* __restrict__ rew,
                                                   float* __restrict__ out_vals,
                                                   int* __restrict__ out_idx) {
  __shared__ float svals[512];
  __shared__ int   sidx[512];
  const int tid = threadIdx.x;
  for (int j = 0; j < NKEEP; ++j) {
    float best = -__builtin_inff();
    int   bidx = 0x7fffffff;
    for (int i = tid; i < NTRY; i += 512) {
      float v = rew[i];
      if (v > best || (v == best && i < bidx)) { best = v; bidx = i; }
    }
    svals[tid] = best; sidx[tid] = bidx;
    __syncthreads();
    for (int off = 256; off > 0; off >>= 1) {
      if (tid < off) {
        float ov = svals[tid + off]; int oi = sidx[tid + off];
        if (ov > svals[tid] || (ov == svals[tid] && oi < sidx[tid])) {
          svals[tid] = ov; sidx[tid] = oi;
        }
      }
      __syncthreads();
    }
    if (tid == 0) {
      out_vals[j] = svals[0];
      out_idx[j]  = sidx[0];
      rew[sidx[0]] = -__builtin_inff();
    }
    __syncthreads();
  }
}

// ---------------------------------------------------------------------------
// Kernel 4: recompute selected actions k_best[t][j][a].
// ---------------------------------------------------------------------------
__global__ void gather_kernel(const float* __restrict__ act_mus,
                              const float* __restrict__ act_sigmas,
                              const float* __restrict__ eps_a,
                              const int* __restrict__ top_idx,
                              float* __restrict__ out) {
  int i = blockIdx.x * blockDim.x + threadIdx.x;
  if (i >= TSTEPS * NKEEP * ADIM) return;
  int a = i % ADIM;
  int j = (i / ADIM) % NKEEP;
  int t = i / (ADIM * NKEEP);
  int idx = top_idx[j];
  out[i] = act_mus[t * ADIM + a] +
           act_sigmas[t * ADIM + a] * eps_a[((size_t)t * NTRY + idx) * ADIM + a];
}

// ---------------------------------------------------------------------------
extern "C" void kernel_launch(void* const* d_in, const int* in_sizes, int n_in,
                              void* d_out, int out_size, void* d_ws, size_t ws_size,
                              hipStream_t stream) {
  const float* h0        = (const float*)d_in[0];
  const float* s0        = (const float*)d_in[1];
  const float* act_mus   = (const float*)d_in[2];
  const float* act_sig   = (const float*)d_in[3];
  const float* eps_a     = (const float*)d_in[4];
  const float* eps_s     = (const float*)d_in[5];
  const float* W_rssm    = (const float*)d_in[6];
  const float* b_rssm    = (const float*)d_in[7];
  const float* W_mu      = (const float*)d_in[8];
  const float* b_mu      = (const float*)d_in[9];
  const float* W_sig     = (const float*)d_in[10];
  const float* b_sig     = (const float*)d_in[11];
  const float* W_rew     = (const float*)d_in[12];
  const float* b_rew     = (const float*)d_in[13];
  float* out = (float*)d_out;

  char* ws = (char*)d_ws;
  size_t off = 0;
  __bf16* Wb_rssm = (__bf16*)(ws + off); off += (size_t)XCOLS * RSSM_N * 2;
  __bf16* Wb_mu   = (__bf16*)(ws + off); off += (size_t)XCOLS * HEAD_N * 2;
  __bf16* Wb_sig  = (__bf16*)(ws + off); off += (size_t)XCOLS * HEAD_N * 2;
  __bf16* Wb_rew  = (__bf16*)(ws + off); off += (size_t)XCOLS * REW_N * 2;
  float*  rew_sum = (float*)(ws + off);  off += (size_t)NTRY * 4;
  int*    top_idx = (int*)(ws + off);    off += NKEEP * 4;

  prep_weights<<<256, 256, 0, stream>>>(W_rssm, W_mu, W_sig, W_rew,
                                        Wb_rssm, Wb_mu, Wb_sig, Wb_rew);

  int blocks = (NTILES + WPB - 1) / WPB;   // 3125
  rollout_kernel<<<blocks, WPB * 32, 0, stream>>>(
      h0, s0, act_mus, act_sig, eps_a, eps_s,
      b_rssm, b_mu, b_sig, b_rew,
      Wb_rssm, Wb_mu, Wb_sig, Wb_rew, rew_sum);

  topk_kernel<<<1, 512, 0, stream>>>(rew_sum, out + TSTEPS * NKEEP * ADIM, top_idx);

  gather_kernel<<<(TSTEPS * NKEEP * ADIM + 255) / 256, 256, 0, stream>>>(
      act_mus, act_sig, eps_a, top_idx, out);
}
